// ChaosAwareTrafficPredictor_50190987821064
// MI455X (gfx1250) — matrix operand order repaired and checked by reading, not compiled, not run on hardware
//
#include <hip/hip_runtime.h>
#include <math.h>

#define NN 640
#define HH 256
#define CC 64
#define PTOT 204480   // 640*639/2

typedef __attribute__((ext_vector_type(16))) _Float16 h16;
typedef __attribute__((ext_vector_type(8)))  _Float16 h8;
typedef __attribute__((ext_vector_type(8)))  float    f8;

__device__ __forceinline__ float gelu_f(float x) {
    return 0.5f * x * (1.0f + erff(x * 0.70710678118654752f));
}
__device__ __forceinline__ float sigmoid_f(float x) {
    return 1.0f / (1.0f + expf(-x));
}

// A-fragment: halves at p[0..7] and p[16..23]  (ISA 16-bit A 16x32 layout)
__device__ __forceinline__ h16 fragA(const _Float16* p) {
    h8 lo = *(const h8*)(p);
    h8 hi = *(const h8*)(p + 16);
    h16 r;
#pragma unroll
    for (int u = 0; u < 8; ++u) { r[u] = lo[u]; r[u + 8] = hi[u]; }
    return r;
}
// B-fragment: 16 contiguous halves  (SWMMAC-style B lane layout)
__device__ __forceinline__ h16 fragB(const _Float16* p) {
    h8 lo = *(const h8*)(p);
    h8 hi = *(const h8*)(p + 8);
    h16 r;
#pragma unroll
    for (int u = 0; u < 8; ++u) { r[u] = lo[u]; r[u + 8] = hi[u]; }
    return r;
}

// ---- block reductions (blockDim.x == 256) ----
__device__ __forceinline__ float block_sum(float v, float* sbuf) {
    int t = threadIdx.x;
    sbuf[t] = v; __syncthreads();
    for (int s = 128; s > 0; s >>= 1) {
        if (t < s) sbuf[t] += sbuf[t + s];
        __syncthreads();
    }
    float r = sbuf[0]; __syncthreads();
    return r;
}
__device__ __forceinline__ float block_max(float v, float* sbuf) {
    int t = threadIdx.x;
    sbuf[t] = v; __syncthreads();
    for (int s = 128; s > 0; s >>= 1) {
        if (t < s) sbuf[t] = fmaxf(sbuf[t], sbuf[t + s]);
        __syncthreads();
    }
    float r = sbuf[0]; __syncthreads();
    return r;
}

// ---------------------------------------------------------------------------
// f32 -> f16 convert
// ---------------------------------------------------------------------------
__global__ void k_cvt(const float* __restrict__ src, _Float16* __restrict__ dst, int n) {
    int i = blockIdx.x * blockDim.x + threadIdx.x;
    if (i < n) dst[i] = (_Float16)src[i];
}

// ---------------------------------------------------------------------------
// init: edge score matrix = 1.0, dist-max = 0
// ---------------------------------------------------------------------------
__global__ void k_init(float* __restrict__ esc, unsigned* __restrict__ dmax, int n) {
    int i = blockIdx.x * blockDim.x + threadIdx.x;
    if (i < n) esc[i] = 1.0f;
    if (i == 0) dmax[0] = 0u;
}

// ---------------------------------------------------------------------------
// Generic WMMA GEMM: out = act(scale * (X @ W^T) + bias)
//   X: [M,K] f16 (ldx), W: [N,K] f16 (ldw). One 32x32 tile per wave
//   (2x2 register blocking: 4 wmma per 8 b128 loads). K mult of 32,
//   M,N mult of 32.
// ---------------------------------------------------------------------------
__device__ __forceinline__ void gemm_epi(f8 acc, int m0, int n0, int lh, int ll,
                                         float scale, const float* bias, int act,
                                         float* oF, int ldoF, _Float16* oH, int ldoH,
                                         _Float16* oHT, int ldoHT) {
    int n = n0 + ll;
    float bv = bias ? bias[n] : 0.0f;
#pragma unroll
    for (int r = 0; r < 8; ++r) {
        int m = m0 + r + (lh << 3);
        float v = acc[r] * scale + bv;
        if (act == 1) v = gelu_f(v);
        if (oF)  oF[(size_t)m * ldoF + n] = v;
        if (oH)  oH[(size_t)m * ldoH + n] = (_Float16)v;
        if (oHT) oHT[(size_t)n * ldoHT + m] = (_Float16)v;
    }
}

__global__ void k_gemm_xt(const _Float16* __restrict__ X, int ldx,
                          const _Float16* __restrict__ Wm, int ldw,
                          int M, int N, int K, float scale,
                          const float* __restrict__ bias, int act,
                          float* __restrict__ oF, int ldoF,
                          _Float16* __restrict__ oH, int ldoH,
                          _Float16* __restrict__ oHT, int ldoHT) {
    int Nt = N >> 5;
    int tiles = (M >> 5) * Nt;
    int tile = blockIdx.x * 8 + (threadIdx.x >> 5);
    if (tile >= tiles) return;
    int m0 = (tile / Nt) << 5;
    int n0 = (tile % Nt) << 5;
    int lane = threadIdx.x & 31;
    int lh = lane >> 4;
    int ll = lane & 15;
    const _Float16* x0 = X + (size_t)(m0 + ll) * ldx;
    const _Float16* x1 = X + (size_t)(m0 + 16 + ll) * ldx;
    const _Float16* w0 = Wm + (size_t)(n0 + ll) * ldw;
    const _Float16* w1 = Wm + (size_t)(n0 + 16 + ll) * ldw;
    __builtin_prefetch(w0, 0, 0);
    __builtin_prefetch(w1, 0, 0);
    int kbA = lh * 8, kbB = lh * 16;

    f8 a00, a01, a10, a11;
#pragma unroll
    for (int u = 0; u < 8; ++u) { a00[u] = 0.0f; a01[u] = 0.0f; a10[u] = 0.0f; a11[u] = 0.0f; }

#pragma unroll 2
    for (int k = 0; k < K; k += 32) {
        h16 fa0 = fragA(x0 + k + kbA);
        h16 fa1 = fragA(x1 + k + kbA);
        h16 fb0 = fragA(w0 + k + kbA);
        h16 fb1 = fragA(w1 + k + kbA);
        a00 = __builtin_amdgcn_wmma_f32_16x16x32_f16(false, fa0, false, fb0, (short)0, a00, false, false);
        a01 = __builtin_amdgcn_wmma_f32_16x16x32_f16(false, fa0, false, fb1, (short)0, a01, false, false);
        a10 = __builtin_amdgcn_wmma_f32_16x16x32_f16(false, fa1, false, fb0, (short)0, a10, false, false);
        a11 = __builtin_amdgcn_wmma_f32_16x16x32_f16(false, fa1, false, fb1, (short)0, a11, false, false);
    }
    gemm_epi(a00, m0,      n0,      lh, ll, scale, bias, act, oF, ldoF, oH, ldoH, oHT, ldoHT);
    gemm_epi(a01, m0,      n0 + 16, lh, ll, scale, bias, act, oF, ldoF, oH, ldoH, oHT, ldoHT);
    gemm_epi(a10, m0 + 16, n0,      lh, ll, scale, bias, act, oF, ldoF, oH, ldoH, oHT, ldoHT);
    gemm_epi(a11, m0 + 16, n0 + 16, lh, ll, scale, bias, act, oF, ldoF, oH, ldoH, oHT, ldoHT);
}

// ---------------------------------------------------------------------------
// small prep: enc_chaos = gelu(ln(chaos@ceW^T+ceb)); apc = ap bias + chaos part;
// cvec = ei_b1 + chaos part of ei_W1. One block of 256 threads.
// ---------------------------------------------------------------------------
__global__ void k_prep(const float* __restrict__ chaos,
                       const float* __restrict__ ceW, const float* __restrict__ ceb,
                       const float* __restrict__ ceg, const float* __restrict__ cebeta,
                       const float* __restrict__ apW, const float* __restrict__ apb,
                       const float* __restrict__ eiW1, const float* __restrict__ eib1,
                       float* __restrict__ apc, float* __restrict__ cvec) {
    __shared__ float sbuf[256];
    __shared__ float encc[256];
    int t = threadIdx.x;
    float z = ceb[t];
    for (int c = 0; c < CC; ++c) z += chaos[c] * ceW[t * CC + c];
    float s = block_sum(z, sbuf);
    float ss = block_sum(z * z, sbuf);
    float mean = s * (1.0f / HH);
    float var = ss * (1.0f / HH) - mean * mean;
    float y = (z - mean) * rsqrtf(var + 1e-5f) * ceg[t] + cebeta[t];
    y = gelu_f(y);
    encc[t] = y;
    __syncthreads();
    float a = apb[t];
    for (int k = 0; k < HH; ++k) a += encc[k] * apW[t * (2 * HH) + HH + k];
    apc[t] = a;
    float cv = eib1[t];
    for (int c = 0; c < CC; ++c) cv += chaos[c] * eiW1[t * (2 * HH + CC) + 2 * HH + c];
    cvec[t] = cv;
}

// ---------------------------------------------------------------------------
// row LayerNorm + GELU  (row length 256), writes f32 + f16
// ---------------------------------------------------------------------------
__global__ void k_ln_gelu(const float* __restrict__ Z, const float* __restrict__ g,
                          const float* __restrict__ b, float* __restrict__ oF,
                          _Float16* __restrict__ oH) {
    __shared__ float sbuf[256];
    int row = blockIdx.x, t = threadIdx.x;
    float x = Z[row * HH + t];
    float s = block_sum(x, sbuf);
    float ss = block_sum(x * x, sbuf);
    float mean = s * (1.0f / HH);
    float var = ss * (1.0f / HH) - mean * mean;
    float y = (x - mean) * rsqrtf(var + 1e-5f) * g[t] + b[t];
    y = gelu_f(y);
    oF[row * HH + t] = y;
    oH[row * HH + t] = (_Float16)y;
}

// ---------------------------------------------------------------------------
// attention softmax over a 640-wide row of S (pre-scaled); writes f16 probs
// ---------------------------------------------------------------------------
__global__ void k_softmax_attn(const float* __restrict__ S, _Float16* __restrict__ Pm) {
    __shared__ float sbuf[256];
    int row = blockIdx.x, t = threadIdx.x;
    float mx = -1e30f;
    for (int j = t; j < NN; j += 256) mx = fmaxf(mx, S[row * NN + j]);
    mx = block_max(mx, sbuf);
    float s = 0.0f;
    for (int j = t; j < NN; j += 256) s += expf(S[row * NN + j] - mx);
    s = block_sum(s, sbuf);
    float inv = 1.0f / s;
    for (int j = t; j < NN; j += 256)
        Pm[row * NN + j] = (_Float16)(expf(S[row * NN + j] - mx) * inv);
}

// ---------------------------------------------------------------------------
// combined = (loc+glo+enc)/3 ; also sq[i] = sum of squares of row
// ---------------------------------------------------------------------------
__global__ void k_combined(const float* __restrict__ loc, const float* __restrict__ glo,
                           const float* __restrict__ enc, float* __restrict__ cF,
                           _Float16* __restrict__ cH, float* __restrict__ sq) {
    __shared__ float sbuf[256];
    int row = blockIdx.x, t = threadIdx.x;
    float v = (loc[row * HH + t] + glo[row * HH + t] + enc[row * HH + t]) * (1.0f / 3.0f);
    cF[row * HH + t] = v;
    cH[row * HH + t] = (_Float16)v;
    float s2 = block_sum(v * v, sbuf);
    if (t == 0) sq[row] = s2;
}

// ---------------------------------------------------------------------------
// global max of pairwise distances (uint-bits atomicMax; dist >= 0)
// ---------------------------------------------------------------------------
__global__ void k_distmax(const float* __restrict__ gram, const float* __restrict__ sq,
                          unsigned* __restrict__ dmax) {
    int idx = blockIdx.x * blockDim.x + threadIdx.x;
    if (idx >= NN * NN) return;
    int i = idx / NN, j = idx % NN;
    float d2 = sq[i] + sq[j] - 2.0f * gram[idx];
    float dist = sqrtf(fmaxf(d2, 0.0f));
    atomicMax(dmax, __float_as_uint(dist));
}

// ---------------------------------------------------------------------------
// combined adjacency: static + sim + distance-MLP + chaos
// ---------------------------------------------------------------------------
__global__ void k_adj(const float* __restrict__ gram, const float* __restrict__ sq,
                      const unsigned* __restrict__ dmax,
                      const float* __restrict__ dmW1, const float* __restrict__ dmb1,
                      const float* __restrict__ dmW2, const float* __restrict__ dmb2,
                      const float* __restrict__ chlog, float* __restrict__ cadj) {
    int idx = blockIdx.x * blockDim.x + threadIdx.x;
    if (idx >= NN * NN) return;
    int i = idx / NN, j = idx % NN;
    float sa = (i != j) ? 1.0f : 0.0f;
    float g = gram[idx];
    float sim = sigmoid_f(g) - ((i == j) ? 1.0f : 0.0f);
    float d2 = sq[i] + sq[j] - 2.0f * g;
    float dist = sqrtf(fmaxf(d2, 0.0f));
    float nd = dist / (__uint_as_float(dmax[0]) + 1e-8f);
    float acc = dmb2[0];
#pragma unroll 4
    for (int c = 0; c < CC; ++c) {
        float h1 = fmaxf(nd * dmW1[c] + dmb1[c], 0.0f);
        acc += h1 * dmW2[c];
    }
    float dw = sigmoid_f(acc);
    float ch = sigmoid_f(chlog[idx]);
    cadj[idx] = 0.3f * sa + 0.25f * sim + 0.25f * dw * sa + 0.2f * ch * sa;
}

// ---------------------------------------------------------------------------
// Fused edge MLP: 64 pairs per block (256 thr = 8 waves).
//  pre = A[i]+B[j]+cvec -> LN(ei_g,ei_beta) -> GELU -> f16 LDS tile [64,256]
//  layer2: ei_W2 fragments held in REGISTERS per wave (8 k-frags), 4 sub-tiles
//  of 16 pairs each: 8 WMMA per sub-tile with A from LDS (ds_load_b128).
//  layer3: GELU -> dot ei_W3 via shuffle-tree + LDS atomics -> sigmoid ->
//  masked scatter into edge score matrix (both triangles).
// ---------------------------------------------------------------------------
__device__ __forceinline__ long long triOff(int a) {
    return (long long)a * NN - (long long)a * (a + 1) / 2;
}

__global__ void k_edge(const float* __restrict__ Af, const float* __restrict__ Bf,
                       const float* __restrict__ cvec,
                       const float* __restrict__ eig, const float* __restrict__ eibeta,
                       const _Float16* __restrict__ W2h, const float* __restrict__ b2,
                       const float* __restrict__ W3, const float* __restrict__ b3,
                       const float* __restrict__ cadj, float* __restrict__ esc) {
    __shared__ _Float16 Xs[64 * 256];   // 32 KB
    __shared__ float acc64[64];
    __shared__ int ii[64], jj[64];
    int t = threadIdx.x;

    // one-shot pair decode for the 64 pairs of this block
    if (t < 64) {
        acc64[t] = 0.0f;
        int p = blockIdx.x * 64 + t;
        int i = 0, j = 0;
        if (p < PTOT) {
            double tN = 2.0 * NN - 1.0;
            double sq_ = sqrt(tN * tN - 8.0 * (double)p);
            int a = (int)((tN - sq_) * 0.5);
            if (a < 0) a = 0;
            if (a > NN - 2) a = NN - 2;
            while (a > 0 && triOff(a) > (long long)p) --a;
            while (triOff(a + 1) <= (long long)p) ++a;
            i = a;
            j = i + 1 + (int)((long long)p - triOff(a));
        }
        ii[t] = i; jj[t] = j;
    }
    __syncthreads();

    // stage 1: 16 threads per pair, 4 groups of 16 pairs
    int pp = t >> 4;
    int l16 = t & 15;
    int kb = l16 * 16;
    for (int g = 0; g < 4; ++g) {
        int slot = g * 16 + pp;
        int p = blockIdx.x * 64 + slot;
        int i = ii[slot], j = jj[slot];
        float pre[16];
        float lsum = 0.0f, lss = 0.0f;
#pragma unroll
        for (int u = 0; u < 16; ++u) {
            float x = 0.0f;
            if (p < PTOT) x = Af[i * HH + kb + u] + Bf[j * HH + kb + u] + cvec[kb + u];
            pre[u] = x; lsum += x; lss += x * x;
        }
#pragma unroll
        for (int m = 8; m >= 1; m >>= 1) {
            lsum += __shfl_xor(lsum, m, 32);
            lss  += __shfl_xor(lss,  m, 32);
        }
        float mean = lsum * (1.0f / HH);
        float var = lss * (1.0f / HH) - mean * mean;
        float rstd = rsqrtf(var + 1e-5f);
#pragma unroll
        for (int u = 0; u < 16; ++u) {
            int k = kb + u;
            float y = (pre[u] - mean) * rstd * eig[k] + eibeta[k];
            Xs[slot * 256 + k] = (_Float16)gelu_f(y);
        }
    }
    __syncthreads();

    // layer 2: wave w -> output cols [w*16, w*16+16); B held in registers
    int lane = t & 31;
    int lh = lane >> 4, ll = lane & 15;
    int n0 = (t >> 5) << 4;
    int n = n0 + ll;
    const _Float16* wrow = W2h + (size_t)n * 256;
    __builtin_prefetch(wrow, 0, 0);
    int kbA = lh * 8, kbB = lh * 16;
    h16 bf[8];
#pragma unroll
    for (int ks = 0; ks < 8; ++ks) bf[ks] = fragA(wrow + ks * 32 + kbA);

    float w3 = W3[n], bb2 = b2[n];
#pragma unroll
    for (int s4 = 0; s4 < 4; ++s4) {
        const _Float16* xrow = Xs + (s4 * 16 + ll) * 256;
        f8 acc;
#pragma unroll
        for (int u = 0; u < 8; ++u) acc[u] = 0.0f;
#pragma unroll
        for (int ks = 0; ks < 8; ++ks) {
            h16 a = fragA(xrow + ks * 32 + kbA);
            acc = __builtin_amdgcn_wmma_f32_16x16x32_f16(false, a, false, bf[ks], (short)0, acc, false, false);
        }
        // layer-3 partial: per-row dot with W3 over this wave's 16 cols
        float hv[8];
#pragma unroll
        for (int r = 0; r < 8; ++r) hv[r] = gelu_f(acc[r] + bb2) * w3;
#pragma unroll
        for (int m = 8; m >= 1; m >>= 1) {
#pragma unroll
            for (int r = 0; r < 8; ++r) hv[r] += __shfl_xor(hv[r], m, 32);
        }
        if (ll == 0) {
#pragma unroll
            for (int r = 0; r < 8; ++r)
                atomicAdd(&acc64[s4 * 16 + r + (lh << 3)], hv[r]);
        }
    }
    __syncthreads();

    if (t < 64) {
        int p2 = blockIdx.x * 64 + t;
        if (p2 < PTOT) {
            int i2 = ii[t], j2 = jj[t];
            float imp = sigmoid_f(acc64[t] + b3[0]);
            float val = (cadj[i2 * NN + j2] > 0.1f) ? imp : 1.0f;
            esc[i2 * NN + j2] = val;
            esc[j2 * NN + i2] = val;
        }
    }
}

// ---------------------------------------------------------------------------
// edge softmax (row-wise, with temperature from edge_temperature input)
// ---------------------------------------------------------------------------
__global__ void k_softmax_edge(const float* __restrict__ esc, const float* __restrict__ et,
                               float* __restrict__ es) {
    __shared__ float sbuf[256];
    int row = blockIdx.x, t = threadIdx.x;
    float invT = 1.0f / (sigmoid_f(et[0]) + 0.1f);
    float mx = -1e30f;
    for (int j = t; j < NN; j += 256) mx = fmaxf(mx, esc[row * NN + j] * invT);
    mx = block_max(mx, sbuf);
    float s = 0.0f;
    for (int j = t; j < NN; j += 256) s += expf(esc[row * NN + j] * invT - mx);
    s = block_sum(s, sbuf);
    float inv = 1.0f / s;
    for (int j = t; j < NN; j += 256)
        es[row * NN + j] = expf(esc[row * NN + j] * invT - mx) * inv;
}

__global__ void k_prod(const float* __restrict__ a, const float* __restrict__ b,
                       float* __restrict__ o, int n) {
    int i = blockIdx.x * blockDim.x + threadIdx.x;
    if (i < n) o[i] = a[i] * b[i];
}

// symmetrize + eye*0.5 + row-normalize -> d_out
__global__ void k_final(const float* __restrict__ f, float* __restrict__ out) {
    __shared__ float sbuf[256];
    int i = blockIdx.x, t = threadIdx.x;
    float s = 0.0f;
    for (int j = t; j < NN; j += 256) {
        float g = 0.5f * (f[i * NN + j] + f[j * NN + i]) + ((i == j) ? 0.5f : 0.0f);
        s += g;
    }
    float tot = block_sum(s, sbuf);
    float inv = 1.0f / (tot + 1e-8f);
    for (int j = t; j < NN; j += 256) {
        float g = 0.5f * (f[i * NN + j] + f[j * NN + i]) + ((i == j) ? 0.5f : 0.0f);
        out[i * NN + j] = g * inv;
    }
}

// ---------------------------------------------------------------------------
// host
// ---------------------------------------------------------------------------
static void gemm(hipStream_t s, const _Float16* X, int ldx, const _Float16* W, int ldw,
                 int M, int N, int K, float scale, const float* bias, int act,
                 float* oF, int ldoF, _Float16* oH, int ldoH, _Float16* oHT, int ldoHT) {
    int tiles = (M / 32) * (N / 32);
    int blocks = (tiles + 7) / 8;
    k_gemm_xt<<<blocks, 256, 0, s>>>(X, ldx, W, ldw, M, N, K, scale, bias, act,
                                     oF, ldoF, oH, ldoH, oHT, ldoHT);
}

extern "C" void kernel_launch(void* const* d_in, const int* in_sizes, int n_in,
                              void* d_out, int out_size, void* d_ws, size_t ws_size,
                              hipStream_t stream) {
    (void)in_sizes; (void)n_in; (void)out_size; (void)ws_size;
    const float* in_node   = (const float*)d_in[0];
    const float* in_chaos  = (const float*)d_in[1];
    const float* ne_W = (const float*)d_in[2];  const float* ne_b = (const float*)d_in[3];
    const float* ne_g = (const float*)d_in[4];  const float* ne_be = (const float*)d_in[5];
    const float* ce_W = (const float*)d_in[6];  const float* ce_b = (const float*)d_in[7];
    const float* ce_g = (const float*)d_in[8];  const float* ce_be = (const float*)d_in[9];
    const float* la_Wq = (const float*)d_in[10]; const float* la_Wk = (const float*)d_in[11];
    const float* la_Wv = (const float*)d_in[12];
    const float* la_bq = (const float*)d_in[13]; const float* la_bk = (const float*)d_in[14];
    const float* la_bv = (const float*)d_in[15];
    const float* la_Wo = (const float*)d_in[16]; const float* la_bo = (const float*)d_in[17];
    const float* ga_Wq = (const float*)d_in[18]; const float* ga_Wk = (const float*)d_in[19];
    const float* ga_Wv = (const float*)d_in[20];
    const float* ga_bq = (const float*)d_in[21]; const float* ga_bk = (const float*)d_in[22];
    const float* ga_bv = (const float*)d_in[23];
    const float* ga_Wo = (const float*)d_in[24]; const float* ga_bo = (const float*)d_in[25];
    const float* dm_W1 = (const float*)d_in[26]; const float* dm_b1 = (const float*)d_in[27];
    const float* dm_W2 = (const float*)d_in[28]; const float* dm_b2 = (const float*)d_in[29];
    const float* ap_W = (const float*)d_in[30];  const float* ap_b = (const float*)d_in[31];
    const float* pred_W = (const float*)d_in[32]; const float* pred_b = (const float*)d_in[33];
    const float* ei_W1 = (const float*)d_in[34]; const float* ei_b1 = (const float*)d_in[35];
    const float* ei_g = (const float*)d_in[36];  const float* ei_be = (const float*)d_in[37];
    const float* ei_W2 = (const float*)d_in[38]; const float* ei_b2 = (const float*)d_in[39];
    const float* ei_W3 = (const float*)d_in[40]; const float* ei_b3 = (const float*)d_in[41];
    const float* e_temp = (const float*)d_in[42];
    float* out = (float*)d_out;

    // ---- workspace arena ----
    char* base = (char*)d_ws;
    size_t off = 0;
    auto alloc = [&](size_t bytes) -> void* {
        void* p = base + off;
        off = (off + bytes + 255) & ~(size_t)255;
        return p;
    };
    _Float16* hNode  = (_Float16*)alloc(NN * 128 * 2);
    _Float16* hNeW   = (_Float16*)alloc(HH * 128 * 2);
    _Float16* hLaWq  = (_Float16*)alloc(HH * HH * 2);
    _Float16* hLaWk  = (_Float16*)alloc(HH * HH * 2);
    _Float16* hLaWv  = (_Float16*)alloc(HH * HH * 2);
    _Float16* hLaWo  = (_Float16*)alloc(HH * HH * 2);
    _Float16* hGaWq  = (_Float16*)alloc(HH * HH * 2);
    _Float16* hGaWk  = (_Float16*)alloc(HH * HH * 2);
    _Float16* hGaWv  = (_Float16*)alloc(HH * HH * 2);
    _Float16* hGaWo  = (_Float16*)alloc(HH * HH * 2);
    _Float16* hApW   = (_Float16*)alloc(HH * 2 * HH * 2);
    _Float16* hPredW = (_Float16*)alloc(NN * HH * 2);
    _Float16* hEiW1  = (_Float16*)alloc(HH * (2 * HH + CC) * 2);
    _Float16* hEiW2  = (_Float16*)alloc(128 * HH * 2);
    float* apc   = (float*)alloc(HH * 4);
    float* cvec  = (float*)alloc(HH * 4);
    float* encZ  = (float*)alloc(NN * HH * 4);
    float* encN  = (float*)alloc(NN * HH * 4);
    _Float16* encNh = (_Float16*)alloc(NN * HH * 2);
    _Float16* Qh = (_Float16*)alloc(NN * HH * 2);
    _Float16* Kh = (_Float16*)alloc(NN * HH * 2);
    _Float16* Vt = (_Float16*)alloc(HH * NN * 2);
    _Float16* Oh = (_Float16*)alloc(NN * HH * 2);
    float* Sf = (float*)alloc((size_t)NN * NN * 4);
    _Float16* Ph = (_Float16*)alloc((size_t)NN * NN * 2);
    float* locF = (float*)alloc(NN * HH * 4);
    float* gloF = (float*)alloc(NN * HH * 4);
    float* combF = (float*)alloc(NN * HH * 4);
    _Float16* combH = (_Float16*)alloc(NN * HH * 2);
    float* sq = (float*)alloc(NN * 4);
    float* gram = (float*)alloc((size_t)NN * NN * 4);
    _Float16* projH = (_Float16*)alloc(NN * HH * 2);
    float* chlog = (float*)alloc((size_t)NN * NN * 4);
    float* Af = (float*)alloc(NN * HH * 4);
    float* Bf = (float*)alloc(NN * HH * 4);
    unsigned* dmax = (unsigned*)alloc(256);
    float* cadj = (float*)alloc((size_t)NN * NN * 4);
    float* esc  = (float*)alloc((size_t)NN * NN * 4);
    float* es   = (float*)alloc((size_t)NN * NN * 4);
    float* fpr  = (float*)alloc((size_t)NN * NN * 4);

    const int TB = 256;
    auto cvt = [&](const float* s, _Float16* d, int n) {
        k_cvt<<<(n + TB - 1) / TB, TB, 0, stream>>>(s, d, n);
    };

    // init + converts
    k_init<<<(NN * NN + TB - 1) / TB, TB, 0, stream>>>(esc, dmax, NN * NN);
    cvt(in_node, hNode, NN * 128);
    cvt(ne_W, hNeW, HH * 128);
    cvt(la_Wq, hLaWq, HH * HH); cvt(la_Wk, hLaWk, HH * HH);
    cvt(la_Wv, hLaWv, HH * HH); cvt(la_Wo, hLaWo, HH * HH);
    cvt(ga_Wq, hGaWq, HH * HH); cvt(ga_Wk, hGaWk, HH * HH);
    cvt(ga_Wv, hGaWv, HH * HH); cvt(ga_Wo, hGaWo, HH * HH);
    cvt(ap_W, hApW, HH * 2 * HH);
    cvt(pred_W, hPredW, NN * HH);
    cvt(ei_W1, hEiW1, HH * (2 * HH + CC));
    cvt(ei_W2, hEiW2, 128 * HH);

    // small prep (enc_chaos folded into apc / cvec)
    k_prep<<<1, 256, 0, stream>>>(in_chaos, ce_W, ce_b, ce_g, ce_be,
                                  ap_W, ap_b, ei_W1, ei_b1, apc, cvec);

    // node encoder
    gemm(stream, hNode, 128, hNeW, 128, NN, HH, 128, 1.0f, ne_b, 0,
         encZ, HH, nullptr, 0, nullptr, 0);
    k_ln_gelu<<<NN, 256, 0, stream>>>(encZ, ne_g, ne_be, encN, encNh);

    // two attentions
    struct AttCfg {
        const _Float16 *Wq, *Wk, *Wv, *Wo;
        const float *bq, *bk, *bv, *bo;
        int nh; float* outF;
    } atts[2] = {
        { hLaWq, hLaWk, hLaWv, hLaWo, la_bq, la_bk, la_bv, la_bo, 4, locF },
        { hGaWq, hGaWk, hGaWv, hGaWo, ga_bq, ga_bk, ga_bv, ga_bo, 2, gloF },
    };
    for (int a = 0; a < 2; ++a) {
        AttCfg& A = atts[a];
        int dh = HH / A.nh;
        float scale = 1.0f / sqrtf((float)dh);
        gemm(stream, encNh, HH, A.Wq, HH, NN, HH, HH, 1.0f, A.bq, 0,
             nullptr, 0, Qh, HH, nullptr, 0);
        gemm(stream, encNh, HH, A.Wk, HH, NN, HH, HH, 1.0f, A.bk, 0,
             nullptr, 0, Kh, HH, nullptr, 0);
        gemm(stream, encNh, HH, A.Wv, HH, NN, HH, HH, 1.0f, A.bv, 0,
             nullptr, 0, nullptr, 0, Vt, NN);   // transposed: Vt[d][m]
        for (int h = 0; h < A.nh; ++h) {
            gemm(stream, Qh + h * dh, HH, Kh + h * dh, HH, NN, NN, dh, scale,
                 nullptr, 0, Sf, NN, nullptr, 0, nullptr, 0);
            k_softmax_attn<<<NN, 256, 0, stream>>>(Sf, Ph);
            gemm(stream, Ph, NN, Vt + (size_t)h * dh * NN, NN, NN, dh, NN, 1.0f,
                 nullptr, 0, nullptr, 0, Oh + h * dh, HH, nullptr, 0);
        }
        gemm(stream, Oh, HH, A.Wo, HH, NN, HH, HH, 1.0f, A.bo, 0,
             A.outF, HH, nullptr, 0, nullptr, 0);
    }

    // combined + gram + chaos projection + edge layer-1 splits
    k_combined<<<NN, 256, 0, stream>>>(locF, gloF, encN, combF, combH, sq);
    gemm(stream, combH, HH, combH, HH, NN, NN, HH, 1.0f, nullptr, 0,
         gram, NN, nullptr, 0, nullptr, 0);
    gemm(stream, combH, HH, hApW, 2 * HH, NN, HH, HH, 1.0f, apc, 1,
         nullptr, 0, projH, HH, nullptr, 0);
    gemm(stream, projH, HH, hPredW, HH, NN, NN, HH, 1.0f, pred_b, 0,
         chlog, NN, nullptr, 0, nullptr, 0);
    gemm(stream, combH, HH, hEiW1, 2 * HH + CC, NN, HH, HH, 1.0f, nullptr, 0,
         Af, HH, nullptr, 0, nullptr, 0);
    gemm(stream, combH, HH, hEiW1 + HH, 2 * HH + CC, NN, HH, HH, 1.0f, nullptr, 0,
         Bf, HH, nullptr, 0, nullptr, 0);

    // adjacency blend
    k_distmax<<<(NN * NN + TB - 1) / TB, TB, 0, stream>>>(gram, sq, dmax);
    k_adj<<<(NN * NN + TB - 1) / TB, TB, 0, stream>>>(gram, sq, dmax, dm_W1, dm_b1,
                                                      dm_W2, dm_b2, chlog, cadj);

    // fused edge MLP over all pairs (64 pairs / block)
    int eblocks = (PTOT + 63) / 64;
    k_edge<<<eblocks, 256, 0, stream>>>(Af, Bf, cvec, ei_g, ei_be, hEiW2, ei_b2,
                                        ei_W3, ei_b3, cadj, esc);

    // edge softmax, product, symmetrize+normalize
    k_softmax_edge<<<NN, 256, 0, stream>>>(esc, e_temp, es);
    k_prod<<<(NN * NN + TB - 1) / TB, TB, 0, stream>>>(cadj, es, fpr, NN * NN);
    k_final<<<NN, 256, 0, stream>>>(fpr, out);
}
